// AttnGraphSAGE_9818295239065
// MI455X (gfx1250) — compile-verified, matrix operand order checked by reading, not run on hardware
//
#include <hip/hip_runtime.h>
#include <hip/hip_bf16.h>

// ---------------------------------------------------------------------------
// AttnGraphSAGE on MI455X (gfx1250, wave32, WMMA)
//   layer: hf = x @ [Wr | Wn | Wa]  (fused 320-wide GEMM, v_wmma_f32_16x16x32_f16,
//          weights pre-swizzled to fragment order, A staged as f16 in LDS,
//          hf padded to 16-row multiple -> unguarded immediate-offset stores)
//          s[n,h] = leaky(x_iq).av_q + leaky(x_jm).av_m      (per node)
//          segment softmax over dst (ordered-int atomicMax, atomicAdd denom)
//          aggr[dst] += sum_h alpha[e,h] * x_jm[src,h,:]
//          h = relu(BN(x_root + aggr))
//   head:  out = h @ head_W + head_b
// ---------------------------------------------------------------------------

typedef __attribute__((ext_vector_type(16))) _Float16 v16h;
typedef __attribute__((ext_vector_type(8)))  _Float16 v8h;
typedef __attribute__((ext_vector_type(4)))  _Float16 v4h;
typedef __attribute__((ext_vector_type(8)))  float    v8f;

#define LEAKY 0.2f
#define HID   64
#define MTOT  320           // 64 (root) + 128 (jm) + 128 (iq)
#define NCT   (MTOT / 16)   // 20 column tiles

static __device__ __forceinline__ int orderFloat(float f) {
  int i = __float_as_int(f);
  return (i >= 0) ? i : (i ^ 0x7fffffff);
}
static __device__ __forceinline__ float unorderFloat(int i) {
  return __int_as_float((i >= 0) ? i : (i ^ 0x7fffffff));
}

// ---------------- utility fills ----------------
__global__ void fill_f32(float* p, float v, int n) {
  int i = blockIdx.x * blockDim.x + threadIdx.x;
  if (i < n) p[i] = v;
}
__global__ void fill_i32(int* p, int v, int n) {
  int i = blockIdx.x * blockDim.x + threadIdx.x;
  if (i < n) p[i] = v;
}

// ---------------- pack [Wr|Wn|Wa] -> f16 in WMMA B-fragment order -------------
// Wfrag[((ct*KB + kb)*32 + lane)*16 + j] = W[k][col]
//   col = ct*16 + lane%16,  k = kb*32 + ((j<8)?j:8+j) + 8*(lane>>4)
__global__ void pack_wfrag(const float* __restrict__ Wr, const float* __restrict__ Wn,
                           const float* __restrict__ Wa, _Float16* __restrict__ Wfrag,
                           int K) {
  int KB = K >> 5;
  int total = NCT * KB * 512;
  int idx = blockIdx.x * blockDim.x + threadIdx.x;
  if (idx >= total) return;
  int j    = idx & 15;
  int lane = (idx >> 4) & 31;
  int blk  = idx >> 9;            // ct*KB + kb
  int kb   = blk % KB;
  int ct   = blk / KB;
  int col  = ct * 16 + (lane & 15);
  int hi   = lane >> 4;
  int k    = kb * 32 + ((j < 8) ? j : (8 + j)) + 8 * hi;
  float v;
  if (col < 64)       v = Wr[k * 64 + col];
  else if (col < 192) v = Wn[k * 128 + (col - 64)];
  else                v = Wa[k * 128 + (col - 192)];
  Wfrag[idx] = (_Float16)v;
}

// ---------------- WMMA GEMM: C[Npad x 320] = A[N x K] (f32) * Wfrag (f16) -----
// block = 256 threads = 8 waves; wave w -> col tile ct = blockIdx.y*8 + w
template<int K>
__global__ void wmma_gemm(const float* __restrict__ A, const _Float16* __restrict__ Wfrag,
                          float* __restrict__ C, int N) {
  constexpr int KPH = K + 8;      // halves; byte stride mult of 16, 4 banks mod 64
  constexpr int KB  = K / 32;
  __shared__ _Float16 As[16 * KPH];
  const int rowBase = blockIdx.x * 16;
  const int tid = threadIdx.x;

  // stage 16 x K A-tile into LDS as f16 (float4 in, v4h out; coalesced)
  for (int i = tid * 4; i < 16 * K; i += 256 * 4) {
    int r = i / K, k = i % K;
    int row = rowBase + r;
    float4 v = make_float4(0.f, 0.f, 0.f, 0.f);
    if (row < N) v = *(const float4*)(A + (size_t)row * K + k);
    v4h h4;
    h4[0] = (_Float16)v.x; h4[1] = (_Float16)v.y;
    h4[2] = (_Float16)v.z; h4[3] = (_Float16)v.w;
    *(v4h*)&As[r * KPH + k] = h4;
  }
  __syncthreads();

  const int wave   = tid >> 5;
  const int lane   = tid & 31;
  const int laneLo = lane & 15;
  const int hi     = lane >> 4;
  const int ct     = blockIdx.y * 8 + wave;
  if (ct >= NCT) return;          // wave-uniform: EXEC all-ones inside
  const int col = ct * 16 + laneLo;

  const _Float16* wp = Wfrag + ((size_t)(ct * KB) * 32 + lane) * 16;

  v8f acc = {};
#pragma unroll
  for (int kb = 0; kb < KB; ++kb) {
    v16h b = *(const v16h*)(wp + (size_t)kb * 512);          // 2x global b128
    const _Float16* ap = &As[laneLo * KPH + kb * 32 + 8 * hi];
    v8h alo = *(const v8h*)ap;                               // ds_load_b128
    v8h ahi = *(const v8h*)(ap + 16);                        // ds_load_b128
    v16h a = __builtin_shufflevector(alo, ahi,
                                     0, 1, 2, 3, 4, 5, 6, 7,
                                     8, 9, 10, 11, 12, 13, 14, 15);
    acc = __builtin_amdgcn_wmma_f32_16x16x32_f16(
        /*neg_a=*/false, a, /*neg_b=*/false, b,
        /*c_mod=*/(short)0, acc, /*reuse_a=*/false, /*reuse_b=*/false);
  }

  // C is padded to 16-row multiple: unguarded stores, constant row offsets
  float* cp = C + (size_t)(rowBase + 8 * hi) * MTOT + col;
#pragma unroll
  for (int r = 0; r < 8; ++r)     // C/D layout: vgpr r -> row rowBase + 8*hi + r
    cp[r * MTOT] = acc[r];
}

// ---------------- per-node attention logits (vectorized b128 loads) ----------
__global__ void score_kernel(const float* __restrict__ hf, const float* __restrict__ av,
                             float* __restrict__ s, int N) {
  int n = blockIdx.x * blockDim.x + threadIdx.x;
  if (n >= N) return;
  const float4* jm = (const float4*)(hf + (size_t)n * MTOT + 64);
  const float4* iq = (const float4*)(hf + (size_t)n * MTOT + 192);
  const float4* avp = (const float4*)av;     // av: 2 x 128 = 64 float4
#pragma unroll
  for (int h = 0; h < 2; ++h) {
    float acc = 0.f;
#pragma unroll 4
    for (int f4 = 0; f4 < 16; ++f4) {
      float4 q = iq[h * 16 + f4];
      float4 m = jm[h * 16 + f4];
      float4 aq = avp[h * 32 + f4];          // av_q
      float4 am = avp[h * 32 + 16 + f4];     // av_m
      acc = fmaf((q.x > 0.f) ? q.x : LEAKY * q.x, aq.x, acc);
      acc = fmaf((q.y > 0.f) ? q.y : LEAKY * q.y, aq.y, acc);
      acc = fmaf((q.z > 0.f) ? q.z : LEAKY * q.z, aq.z, acc);
      acc = fmaf((q.w > 0.f) ? q.w : LEAKY * q.w, aq.w, acc);
      acc = fmaf((m.x > 0.f) ? m.x : LEAKY * m.x, am.x, acc);
      acc = fmaf((m.y > 0.f) ? m.y : LEAKY * m.y, am.y, acc);
      acc = fmaf((m.z > 0.f) ? m.z : LEAKY * m.z, am.z, acc);
      acc = fmaf((m.w > 0.f) ? m.w : LEAKY * m.w, am.w, acc);
    }
    s[n * 2 + h] = acc;
  }
}

// ---------------- segment max over dst (ordered-int atomicMax) ----------------
__global__ void edge_max(const int* __restrict__ src, const int* __restrict__ dst,
                         const float* __restrict__ s, int* __restrict__ m, int E) {
  int e = blockIdx.x * blockDim.x + threadIdx.x;
  if (e >= E) return;
  int sN = src[e], dN = dst[e];
  atomicMax(&m[dN * 2 + 0], orderFloat(s[sN * 2 + 0]));
  atomicMax(&m[dN * 2 + 1], orderFloat(s[sN * 2 + 1]));
}

// ---------------- e = exp(att - max), denom += e ----------------
__global__ void edge_exp(const int* __restrict__ src, const int* __restrict__ dst,
                         const float* __restrict__ s, const int* __restrict__ m,
                         float* __restrict__ ebuf, float* __restrict__ denom, int E) {
  int e = blockIdx.x * blockDim.x + threadIdx.x;
  if (e >= E) return;
  int sN = src[e], dN = dst[e];
#pragma unroll
  for (int h = 0; h < 2; ++h) {
    float ex = __expf(s[sN * 2 + h] - unorderFloat(m[dN * 2 + h]));
    ebuf[e * 2 + h] = ex;
    atomicAdd(&denom[dN * 2 + h], ex);
  }
}

// ---------------- aggr init = x_root (float4) ----------------
__global__ void init_aggr(const float* __restrict__ hf, float* __restrict__ aggr, int N) {
  int idx = blockIdx.x * blockDim.x + threadIdx.x;   // N * 16 float4s
  if (idx >= N * 16) return;
  int n = idx >> 4, f4 = idx & 15;
  ((float4*)aggr)[idx] = ((const float4*)hf)[(size_t)n * (MTOT / 4) + f4];
}

// ---------------- gather/scatter aggregation: (edge, feature) parallel ----------
__global__ void edge_aggr(const int* __restrict__ src, const int* __restrict__ dst,
                          const float* __restrict__ hf, const float* __restrict__ ebuf,
                          const float* __restrict__ denom, float* __restrict__ aggr, int E) {
  int idx = blockIdx.x * blockDim.x + threadIdx.x;   // E * 64 threads
  int e = idx >> 6, f = idx & 63;
  if (e >= E) return;
  int sN = src[e], dN = dst[e];
  float a0 = ebuf[e * 2 + 0] / (denom[dN * 2 + 0] + 1e-16f);
  float a1 = ebuf[e * 2 + 1] / (denom[dN * 2 + 1] + 1e-16f);
  const float* jm = hf + (size_t)sN * MTOT + 64;
  float msg = fmaf(a0, jm[f], a1 * jm[64 + f]);
  atomicAdd(&aggr[(size_t)dN * HID + f], msg);
}

// ---------------- BN stats: coalesced partial sums, atomic reduce ----------------
#define BN_ROWS 128
__global__ void bn_stats(const float* __restrict__ x, float* __restrict__ sums, int N) {
  __shared__ float sh0[256], sh1[256];
  int col = threadIdx.x & 63;
  int g   = threadIdx.x >> 6;                 // 4 row-groups
  int r0  = blockIdx.x * BN_ROWS;
  int r1  = min(N, r0 + BN_ROWS);
  float s0 = 0.f, s1 = 0.f;
  for (int r = r0 + g; r < r1; r += 4) {
    float v = x[(size_t)r * HID + col];
    s0 += v; s1 += v * v;
  }
  sh0[threadIdx.x] = s0; sh1[threadIdx.x] = s1;
  __syncthreads();
  if (g == 0) {
    s0 = sh0[col] + sh0[col + 64] + sh0[col + 128] + sh0[col + 192];
    s1 = sh1[col] + sh1[col + 64] + sh1[col + 128] + sh1[col + 192];
    atomicAdd(&sums[col], s0);
    atomicAdd(&sums[64 + col], s1);
  }
}

// ---------------- BN apply + ReLU (float4) ----------------
__global__ void bn_apply(const float* __restrict__ x, const float* __restrict__ sums,
                         const float* __restrict__ gamma, const float* __restrict__ beta,
                         float* __restrict__ out, int N) {
  int idx = blockIdx.x * blockDim.x + threadIdx.x;   // N * 16 float4s
  if (idx >= N * 16) return;
  int c0 = (idx & 15) * 4;
  float invN = 1.0f / (float)N;
  float4 xv = ((const float4*)x)[idx];
  float4 r;
  {
    float mu = sums[c0 + 0] * invN, var = sums[64 + c0 + 0] * invN - mu * mu;
    r.x = fmaxf((xv.x - mu) * rsqrtf(var + 1e-5f) * gamma[c0 + 0] + beta[c0 + 0], 0.f);
  }
  {
    float mu = sums[c0 + 1] * invN, var = sums[64 + c0 + 1] * invN - mu * mu;
    r.y = fmaxf((xv.y - mu) * rsqrtf(var + 1e-5f) * gamma[c0 + 1] + beta[c0 + 1], 0.f);
  }
  {
    float mu = sums[c0 + 2] * invN, var = sums[64 + c0 + 2] * invN - mu * mu;
    r.z = fmaxf((xv.z - mu) * rsqrtf(var + 1e-5f) * gamma[c0 + 2] + beta[c0 + 2], 0.f);
  }
  {
    float mu = sums[c0 + 3] * invN, var = sums[64 + c0 + 3] * invN - mu * mu;
    r.w = fmaxf((xv.w - mu) * rsqrtf(var + 1e-5f) * gamma[c0 + 3] + beta[c0 + 3], 0.f);
  }
  ((float4*)out)[idx] = r;
}

// ---------------- head: out = h @ W(64x3) + b (float4 row loads) --------------
__global__ void head_kernel(const float* __restrict__ h, const float* __restrict__ W,
                            const float* __restrict__ b, float* __restrict__ out, int N) {
  int n = blockIdx.x * blockDim.x + threadIdx.x;
  if (n >= N) return;
  float a0 = b[0], a1 = b[1], a2 = b[2];
  const float4* row = (const float4*)(h + (size_t)n * HID);
#pragma unroll 4
  for (int f4 = 0; f4 < 16; ++f4) {
    float4 v = row[f4];
    int f = f4 * 4;
    a0 = fmaf(v.x, W[(f + 0) * 3 + 0], a0);
    a1 = fmaf(v.x, W[(f + 0) * 3 + 1], a1);
    a2 = fmaf(v.x, W[(f + 0) * 3 + 2], a2);
    a0 = fmaf(v.y, W[(f + 1) * 3 + 0], a0);
    a1 = fmaf(v.y, W[(f + 1) * 3 + 1], a1);
    a2 = fmaf(v.y, W[(f + 1) * 3 + 2], a2);
    a0 = fmaf(v.z, W[(f + 2) * 3 + 0], a0);
    a1 = fmaf(v.z, W[(f + 2) * 3 + 1], a1);
    a2 = fmaf(v.z, W[(f + 2) * 3 + 2], a2);
    a0 = fmaf(v.w, W[(f + 3) * 3 + 0], a0);
    a1 = fmaf(v.w, W[(f + 3) * 3 + 1], a1);
    a2 = fmaf(v.w, W[(f + 3) * 3 + 2], a2);
  }
  out[n * 3 + 0] = a0; out[n * 3 + 1] = a1; out[n * 3 + 2] = a2;
}

// ---------------------------------------------------------------------------
static inline int cdiv(long long a, long long b) { return (int)((a + b - 1) / b); }

extern "C" void kernel_launch(void* const* d_in, const int* in_sizes, int n_in,
                              void* d_out, int out_size, void* d_ws, size_t ws_size,
                              hipStream_t stream) {
  // inputs: x, edge_index, Wr0, Wn0, Wa0, av0, g0, b0, Wr1, Wn1, Wa1, av1, g1, b1, head_W, head_b
  const float* x   = (const float*)d_in[0];
  const int*   ei  = (const int*)d_in[1];
  const float* Wr0 = (const float*)d_in[2];
  const float* Wn0 = (const float*)d_in[3];
  const float* Wa0 = (const float*)d_in[4];
  const float* av0 = (const float*)d_in[5];
  const float* g0  = (const float*)d_in[6];
  const float* b0  = (const float*)d_in[7];
  const float* Wr1 = (const float*)d_in[8];
  const float* Wn1 = (const float*)d_in[9];
  const float* Wa1 = (const float*)d_in[10];
  const float* av1 = (const float*)d_in[11];
  const float* g1  = (const float*)d_in[12];
  const float* b1  = (const float*)d_in[13];
  const float* hW  = (const float*)d_in[14];
  const float* hb  = (const float*)d_in[15];

  const int IN   = in_sizes[2] / HID;        // Wr0: IN x HID
  const int N    = in_sizes[0] / IN;
  const int E    = in_sizes[1] / 2;
  const int Npad = (N + 15) & ~15;           // GEMM output rows (unguarded stores)
  const int* src = ei;                       // edge_index row 0
  const int* dst = ei + E;                   // edge_index row 1

  // ---- workspace carve-up (256B aligned slices) ----
  char* w = (char*)d_ws;
  auto take = [&](size_t bytes) -> void* {
    void* p = (void*)w;
    w += (bytes + 255) & ~(size_t)255;
    return p;
  };
  float*    hf     = (float*)take((size_t)Npad * MTOT * 4);  // fused [root|jm|iq]
  float*    aggr   = (float*)take((size_t)N * HID * 4);
  float*    hbuf   = (float*)take((size_t)N * HID * 4);
  float*    sbuf   = (float*)take((size_t)N * 2 * 4);
  int*      mbuf   = (int*)  take((size_t)N * 2 * 4);
  float*    denom  = (float*)take((size_t)N * 2 * 4);
  float*    ebuf   = (float*)take((size_t)E * 2 * 4);
  float*    stats  = (float*)take(128 * 4);
  _Float16* Wfrag0 = (_Float16*)take((size_t)IN * MTOT * 2);
  _Float16* Wfrag1 = (_Float16*)take((size_t)HID * MTOT * 2);

  const int NEG_INF_ORD = (int)0x807FFFFF;   // orderFloat(-inf)

  // ---- pack weights into WMMA fragment order (f16) ----
  pack_wfrag<<<cdiv((long long)IN * MTOT, 256), 256, 0, stream>>>(Wr0, Wn0, Wa0, Wfrag0, IN);
  pack_wfrag<<<cdiv((long long)HID * MTOT, 256), 256, 0, stream>>>(Wr1, Wn1, Wa1, Wfrag1, HID);

  dim3 gemmGrid(Npad / 16, cdiv(NCT, 8));

  // =================== layer 0 ===================
  if (IN == 128) wmma_gemm<128><<<gemmGrid, 256, 0, stream>>>(x, Wfrag0, hf, N);
  else           wmma_gemm< 64><<<gemmGrid, 256, 0, stream>>>(x, Wfrag0, hf, N);

  score_kernel<<<cdiv(N, 256), 256, 0, stream>>>(hf, av0, sbuf, N);
  fill_i32<<<cdiv(N * 2, 256), 256, 0, stream>>>(mbuf, NEG_INF_ORD, N * 2);
  fill_f32<<<cdiv(N * 2, 256), 256, 0, stream>>>(denom, 0.f, N * 2);
  edge_max<<<cdiv(E, 256), 256, 0, stream>>>(src, dst, sbuf, mbuf, E);
  edge_exp<<<cdiv(E, 256), 256, 0, stream>>>(src, dst, sbuf, mbuf, ebuf, denom, E);
  init_aggr<<<cdiv((long long)N * 16, 256), 256, 0, stream>>>(hf, aggr, N);
  edge_aggr<<<cdiv((long long)E * HID, 256), 256, 0, stream>>>(src, dst, hf, ebuf, denom, aggr, E);
  fill_f32<<<1, 128, 0, stream>>>(stats, 0.f, 128);
  bn_stats<<<cdiv(N, BN_ROWS), 256, 0, stream>>>(aggr, stats, N);
  bn_apply<<<cdiv((long long)N * 16, 256), 256, 0, stream>>>(aggr, stats, g0, b0, hbuf, N);

  // =================== layer 1 (K = HID = 64) ===================
  wmma_gemm<64><<<gemmGrid, 256, 0, stream>>>(hbuf, Wfrag1, hf, N);

  score_kernel<<<cdiv(N, 256), 256, 0, stream>>>(hf, av1, sbuf, N);
  fill_i32<<<cdiv(N * 2, 256), 256, 0, stream>>>(mbuf, NEG_INF_ORD, N * 2);
  fill_f32<<<cdiv(N * 2, 256), 256, 0, stream>>>(denom, 0.f, N * 2);
  edge_max<<<cdiv(E, 256), 256, 0, stream>>>(src, dst, sbuf, mbuf, E);
  edge_exp<<<cdiv(E, 256), 256, 0, stream>>>(src, dst, sbuf, mbuf, ebuf, denom, E);
  init_aggr<<<cdiv((long long)N * 16, 256), 256, 0, stream>>>(hf, aggr, N);
  edge_aggr<<<cdiv((long long)E * HID, 256), 256, 0, stream>>>(src, dst, hf, ebuf, denom, aggr, E);
  fill_f32<<<1, 128, 0, stream>>>(stats, 0.f, 128);
  bn_stats<<<cdiv(N, BN_ROWS), 256, 0, stream>>>(aggr, stats, N);
  bn_apply<<<cdiv((long long)N * 16, 256), 256, 0, stream>>>(aggr, stats, g1, b1, hbuf, N);

  // =================== head ===================
  head_kernel<<<cdiv(N, 256), 256, 0, stream>>>(hbuf, hW, hb, (float*)d_out, N);
}
